// MixedDecoder_3908420239533
// MI455X (gfx1250) — matrix operand order repaired and checked by reading, not compiled, not run on hardware
//
#include <hip/hip_runtime.h>
#include <hip/hip_bf16.h>
#include <math.h>

// ---------------------------------------------------------------------------
// MixedDecoder on gfx1250 (MI455X).
// Only the gate MLP and the final expert layer (w2,b2) are live in the
// reference (layers 0/1 outputs are discarded). We fold the softmax mixing
// coefficients into the A-matrix rows (c[b,e]*z[b,k]) so a single f32 WMMA
// accumulator reduces over experts*K = 8*288 with v_wmma_f32_16x16x32_bf16.
// B-panels are staged in LDS via async global->LDS copies and shared by the
// 4 waves of each workgroup. A-scaling is done in bf16 (v_pk_mul_bf16).
// ---------------------------------------------------------------------------

typedef __attribute__((ext_vector_type(16))) __bf16 v16bf;
typedef __attribute__((ext_vector_type(8)))  float  v8f;
typedef __attribute__((ext_vector_type(4)))  int    i4;

#define B_SZ     2048
#define IN_SZ    288      // = 9 * 32 k-steps
#define OUT_SZ   256
#define NE       8
#define GH       64
#define KSTEPS   9        // 288 / 32
#define NGRPS    8        // column groups of 32
#define NT       2        // 16-col tiles per group
#define PANEL_TILES (NE * KSTEPS * NT)        // 144 tiles per column group
#define PANEL_ELEMS (PANEL_TILES * 512)       // 73728 bf16 = 144 KB
#define PANEL_BYTES (PANEL_ELEMS * 2)

// workspace layout (bytes)
#define WS_COEF    0                                   // 2048*8  f32 = 64 KB
#define WS_BIASMIX (WS_COEF + B_SZ*NE*4)               // 2048*256 f32 = 2 MB
#define WS_W2BF    (WS_BIASMIX + B_SZ*OUT_SZ*4)        // 8*288*256 bf16 swizzled
#define WS_ZBF     (WS_W2BF + NE*IN_SZ*OUT_SZ*2)       // 2048*288 bf16

typedef __attribute__((address_space(1))) i4 gi4;   // global int4
typedef __attribute__((address_space(3))) i4 li4;   // LDS int4

// 16-byte global -> LDS copy; async (ASYNCcnt) when the gfx1250 builtin exists.
__device__ __forceinline__ void copy16_g2l(__bf16* l, const __bf16* g) {
#if __has_builtin(__builtin_amdgcn_global_load_async_to_lds_b128)
  __builtin_amdgcn_global_load_async_to_lds_b128(
      (gi4*)g, (li4*)l, /*offset=*/0, /*cpol=*/0);
#else
  *(uint4*)l = *(const uint4*)g;
#endif
}

__device__ __forceinline__ void wait_async0() {
#if __has_builtin(__builtin_amdgcn_s_wait_asynccnt)
  __builtin_amdgcn_s_wait_asynccnt(0);
#else
  asm volatile("s_wait_asynccnt 0x0" ::: "memory");
#endif
}

// ---------------------------------------------------------------------------
// Kernel 1: gate MLP (fp32 VALU) -> coef[b][e], biasmix[b][o] = sum_e c*b2
// One thread per batch row. Weight reads are wave-uniform (L2 broadcast).
// ---------------------------------------------------------------------------
__global__ __launch_bounds__(128)
void gate_kernel(const float* __restrict__ z,
                 const float* __restrict__ g0w, const float* __restrict__ g0b,
                 const float* __restrict__ g1w, const float* __restrict__ g1b,
                 const float* __restrict__ g2w, const float* __restrict__ g2b,
                 const float* __restrict__ b2,
                 float* __restrict__ coef, float* __restrict__ biasmix) {
  int b = blockIdx.x * blockDim.x + threadIdx.x;
  if (b >= B_SZ) return;
  const float* zr = z + b * IN_SZ;

  float h0[GH];
  #pragma unroll
  for (int j = 0; j < GH; ++j) h0[j] = g0b[j];
  for (int k = 0; k < IN_SZ; ++k) {
    float zv = zr[k];
    const float* wr = g0w + k * GH;
    #pragma unroll
    for (int j = 0; j < GH; ++j) h0[j] = fmaf(zv, wr[j], h0[j]);
  }
  #pragma unroll
  for (int j = 0; j < GH; ++j) h0[j] = (h0[j] > 0.f) ? h0[j] : expm1f(h0[j]);

  float h1[GH];
  #pragma unroll
  for (int j = 0; j < GH; ++j) h1[j] = g1b[j];
  for (int k = 0; k < GH; ++k) {
    float hv = h0[k];
    const float* wr = g1w + k * GH;
    #pragma unroll
    for (int j = 0; j < GH; ++j) h1[j] = fmaf(hv, wr[j], h1[j]);
  }
  #pragma unroll
  for (int j = 0; j < GH; ++j) h1[j] = (h1[j] > 0.f) ? h1[j] : expm1f(h1[j]);

  float lg[NE];
  #pragma unroll
  for (int e = 0; e < NE; ++e) lg[e] = g2b[e];
  for (int k = 0; k < GH; ++k) {
    float hv = h1[k];
    const float* wr = g2w + k * NE;
    #pragma unroll
    for (int e = 0; e < NE; ++e) lg[e] = fmaf(hv, wr[e], lg[e]);
  }
  // softmax over E=8
  float mx = lg[0];
  #pragma unroll
  for (int e = 1; e < NE; ++e) mx = fmaxf(mx, lg[e]);
  float sum = 0.f;
  float c[NE];
  #pragma unroll
  for (int e = 0; e < NE; ++e) { c[e] = expf(lg[e] - mx); sum += c[e]; }
  float inv = 1.f / sum;
  #pragma unroll
  for (int e = 0; e < NE; ++e) {
    c[e] *= inv;
    coef[b * NE + e] = c[e];
  }
  // premixed bias: biasmix[b][o] = sum_e c[e] * b2[e][o]
  for (int o = 0; o < OUT_SZ; ++o) {
    float s = 0.f;
    #pragma unroll
    for (int e = 0; e < NE; ++e) s = fmaf(c[e], b2[e * OUT_SZ + o], s);
    biasmix[b * OUT_SZ + o] = s;
  }
}

// ---------------------------------------------------------------------------
// Kernel 2: convert w2 (f32 [E,288,256]) -> bf16 pre-swizzled into WMMA
// B-matrix register order, packed so each column group's panel is contiguous:
//   tile id = ((ngrp*NE + e)*KSTEPS + kt)*NT + nt   (512 bf16 per 32x16 tile)
// within a tile, lane L element i = w2[e][kt*32+(L/16)*16+i][(ngrp*2+nt)*16+L%16].
// ---------------------------------------------------------------------------
__global__ __launch_bounds__(256)
void swz_kernel(const float* __restrict__ w2, __bf16* __restrict__ w2bf) {
  int tid = blockIdx.x * blockDim.x + threadIdx.x;
  if (tid >= NE * IN_SZ * OUT_SZ) return;
  int t    = tid >> 9;        // tile index (0..1151)
  int r    = tid & 511;       // element within tile
  int lane = r >> 4;
  int i    = r & 15;
  int ngrp = t / PANEL_TILES;
  int p    = t % PANEL_TILES; // tile within panel
  int e    = p / (KSTEPS * NT);
  int q    = p % (KSTEPS * NT);
  int kt   = q >> 1;
  int nt   = q & 1;
  int k    = kt * 32 + (lane >> 4) * 16 + i;
  int o    = (ngrp * NT + nt) * 16 + (lane & 15);
  w2bf[tid] = (__bf16)w2[(e * IN_SZ + k) * OUT_SZ + o];
}

// ---------------------------------------------------------------------------
// Kernel 2b: z (f32, row-major [B,288]) -> bf16 same layout.
// ---------------------------------------------------------------------------
__global__ __launch_bounds__(256)
void zb_kernel(const float* __restrict__ z, __bf16* __restrict__ zbf) {
  int tid = blockIdx.x * blockDim.x + threadIdx.x;
  if (tid < B_SZ * IN_SZ) zbf[tid] = (__bf16)z[tid];
}

// ---------------------------------------------------------------------------
// Kernel 3: fused mixture GEMM. 4 waves/block; wave w computes a 16x32 tile
// for batch rows [(blockIdx.x*4+w)*16, +16), cols [ngrp*32, +32).
// B panel (144 KB) async-copied to LDS once, shared by all 4 waves.
// A scaled per expert with v_pk_mul_bf16. 144 WMMAs per wave.
// ---------------------------------------------------------------------------
__global__ __launch_bounds__(128)
void moe_kernel(const __bf16* __restrict__ zbf,
                const float* __restrict__ coef,
                const float* __restrict__ biasmix,
                const __bf16* __restrict__ w2bf,
                float* __restrict__ out) {
  extern __shared__ __bf16 ldsb[];           // PANEL_ELEMS bf16 = 144 KB

  const int tid  = threadIdx.x;
  const int wave = tid >> 5;
  const int lane = tid & 31;
  const int ngrp = blockIdx.y;

  // ---- stage this column group's full B panel into LDS (linear 144 KB copy)
  {
    const __bf16* panel = w2bf + (size_t)ngrp * PANEL_ELEMS;
    #pragma unroll 8
    for (int j = 0; j < PANEL_BYTES / (128 * 16); ++j) {   // 72 chunks/thread
      const int c = j * 128 + tid;                          // 16B chunk id
      copy16_g2l(ldsb + c * 8, panel + c * 8);
    }
  }
  wait_async0();
  __syncthreads();

  const int b0 = (blockIdx.x * 4 + wave) * 16;
  const int m  = lane & 15;
  const int hi = lane >> 4;

  // per-lane bf16 mixing coefficients for row b0+m (A-layout: lane = row m)
  __bf16 cb[NE];
  #pragma unroll
  for (int e = 0; e < NE; ++e) cb[e] = (__bf16)coef[(b0 + m) * NE + e];

  v8f acc[NT];
  #pragma unroll
  for (int nt = 0; nt < NT; ++nt)
    acc[nt] = (v8f){0.f,0.f,0.f,0.f,0.f,0.f,0.f,0.f};

  const __bf16* zrow = zbf + (size_t)(b0 + m) * IN_SZ;

  for (int kt = 0; kt < KSTEPS; ++kt) {
    const int k = kt * 32;
    // A-matrix bf16 fetch (contiguous k order used here: lane L holds row m=L%16,
    // K = (L/16)*16 + [0..15] in elems 0..15, matching B)
    union { v16bf v; uint4 q[2]; } za;
    za.q[0] = *(const uint4*)(zrow + k + hi * 16);
    za.q[1] = *(const uint4*)(zrow + k + hi * 16 + 8);

    #pragma unroll
    for (int e = 0; e < NE; ++e) {
      const v16bf a = za.v * cb[e];     // 8x v_pk_mul_bf16 (splat scale)
      #pragma unroll
      for (int nt = 0; nt < NT; ++nt) {
        const int p = (e * KSTEPS + kt) * NT + nt;       // tile within panel
        const v16bf bv = *(const v16bf*)(ldsb + p * 512 + lane * 16);
        acc[nt] = __builtin_amdgcn_wmma_f32_16x16x32_bf16(
            /*neg_a=*/false, a, /*neg_b=*/false, bv,
            /*c_mod=*/(short)0, acc[nt], /*reuse_a=*/false, /*reuse_b=*/false);
      }
    }
  }

  // epilogue: D layout -> row = r + 8*hi, col = lane%16; add premixed bias
  #pragma unroll
  for (int nt = 0; nt < NT; ++nt) {
    const int o0 = (ngrp * NT + nt) * 16;
    #pragma unroll
    for (int r = 0; r < 8; ++r) {
      const int idx = (b0 + r + hi * 8) * OUT_SZ + (o0 + m);
      out[idx] = acc[nt][r] + biasmix[idx];
    }
  }
}

// ---------------------------------------------------------------------------
extern "C" void kernel_launch(void* const* d_in, const int* in_sizes, int n_in,
                              void* d_out, int out_size, void* d_ws, size_t ws_size,
                              hipStream_t stream) {
  const float* z   = (const float*)d_in[0];
  const float* g0w = (const float*)d_in[1];
  const float* g0b = (const float*)d_in[2];
  const float* g1w = (const float*)d_in[3];
  const float* g1b = (const float*)d_in[4];
  const float* g2w = (const float*)d_in[5];
  const float* g2b = (const float*)d_in[6];
  // d_in[7..10] = w0,b0,w1,b1 : dead in the reference (outputs discarded)
  const float* w2  = (const float*)d_in[11];
  const float* b2  = (const float*)d_in[12];
  float* out = (float*)d_out;

  char* ws = (char*)d_ws;
  float*  coef    = (float*)(ws + WS_COEF);
  float*  biasmix = (float*)(ws + WS_BIASMIX);
  __bf16* w2bf    = (__bf16*)(ws + WS_W2BF);
  __bf16* zbf     = (__bf16*)(ws + WS_ZBF);

  // 1) gating network + premixed bias
  gate_kernel<<<B_SZ / 128, 128, 0, stream>>>(z, g0w, g0b, g1w, g1b, g2w, g2b,
                                              b2, coef, biasmix);
  // 2) w2 -> bf16, WMMA B-register swizzle (panel-contiguous per column group)
  swz_kernel<<<(NE * IN_SZ * OUT_SZ + 255) / 256, 256, 0, stream>>>(w2, w2bf);
  // 2b) z -> bf16
  zb_kernel<<<(B_SZ * IN_SZ + 255) / 256, 256, 0, stream>>>(z, zbf);
  // 3) fused mixture GEMM (WMMA + LDS-staged B)
  moe_kernel<<<dim3(B_SZ / 64, NGRPS), 128, PANEL_BYTES, stream>>>(
      zbf, coef, biasmix, w2bf, out);
}
